// PRLModel_68925635166859
// MI455X (gfx1250) — matrix operand order repaired and checked by reading, not compiled, not run on hardware
//
#include <hip/hip_runtime.h>

#define D_MODEL 512
#define NHEAD 8
#define HDIM 64
#define LSEQ 1024
#define BATCH 8
#define MROWS (BATCH * LSEQ)
#define DFF 2048
#define DUSER 256
#define KEXP 8
#define KCONV 5

typedef __attribute__((ext_vector_type(16))) __bf16 v16bf;
typedef __attribute__((ext_vector_type(8)))  float  v8f;
typedef __attribute__((ext_vector_type(4)))  unsigned int u32x4;
typedef __attribute__((ext_vector_type(4)))  float  f32x4;

union FragAB { v16bf v; u32x4 q[2]; };

// pack two floats into two bf16 (round-half-up) in one dword
__device__ __forceinline__ unsigned pk2(float lo, float hi) {
  unsigned a = (__float_as_uint(lo) + 0x8000u) >> 16;
  unsigned b = (__float_as_uint(hi) + 0x8000u) & 0xFFFF0000u;
  return a | b;
}

__device__ __forceinline__ unsigned short f2bf(float f) {
  return (unsigned short)((__float_as_uint(f) + 0x8000u) >> 16);
}

__device__ __forceinline__ v8f vzero8() {
  v8f z;
  #pragma unroll
  for (int i = 0; i < 8; ++i) z[i] = 0.0f;
  return z;
}

// ---------------------------------------------------------------------------
// bf16 WMMA GEMM: C[M,N] (+)= act( A'[M,K] @ B[K,N] + bias )
//   Requirements (all call sites satisfy): M%64==0, N%128==0 per grid, K%32==0.
//   A' rows optionally shifted by `ashift` within segments of length `lmod`
//   (out-of-range rows contribute zero -> causal conv), optionally scaled
//   per-row by rowScale (delta-weighted V input).
// flags: bit0 = accumulate into C, bit1 = relu
// Double-buffered LDS; register-staged global loads overlap with WMMA.
// ---------------------------------------------------------------------------
__global__ __launch_bounds__(256) void gemm_wmma(
    const float* __restrict__ A, int lda, int ashift, int lmod,
    const float* __restrict__ rowScale,
    const float* __restrict__ B, int ldb,
    float* __restrict__ C, int ldc,
    const float* __restrict__ bias,
    int M, int N, int Kdim, int flags)
{
  __shared__ __attribute__((aligned(16))) unsigned short As[2][64][32];
  __shared__ __attribute__((aligned(16))) unsigned short Bs[2][128][32];

  const int tid  = threadIdx.x;
  const int lane = tid & 31;
  const int wid  = tid >> 5;
  const int hs   = lane >> 4;
  const int l16  = lane & 15;
  const int wm   = wid >> 2;   // 0..1
  const int wn   = wid & 3;    // 0..3
  const int m0   = blockIdx.y * 64;
  const int n0   = blockIdx.x * 128;

  v8f acc[2][2];
  #pragma unroll
  for (int i = 0; i < 2; ++i)
    #pragma unroll
    for (int j = 0; j < 2; ++j) acc[i][j] = vzero8();

  // A tile: thread loads 8 consecutive floats of one row
  const int aRow = tid >> 2;
  const int aCol = (tid & 3) * 8;
  // B tile: thread loads a k-row pair x 8 columns (packs transpose naturally)
  const int bKp  = tid >> 4;        // 0..15 (k pair)
  const int bN8  = (tid & 15) * 8;  // 0..120

  const int mrow = m0 + aRow;
  bool valid = (mrow < M);
  const int arow = mrow + ashift;
  if (lmod) {
    int lp = (mrow % lmod) + ashift;
    valid = valid && (lp >= 0) && (lp < lmod);
  }
  float rsc = 0.0f;
  if (valid) rsc = (rowScale != nullptr) ? rowScale[arow] : 1.0f;
  const float* ap = A + (size_t)(valid ? arow : 0) * lda + aCol;
  const float* bbase = B + n0 + bN8 + (size_t)(2 * bKp) * ldb;

  f32x4 ra0, ra1, rb0, rb1, rb2, rb3;

  const int nk = Kdim >> 5;

  // prologue: load + stage tile 0
  ra0 = *(const f32x4*)(ap + 0);
  ra1 = *(const f32x4*)(ap + 4);
  rb0 = *(const f32x4*)(bbase + 0);
  rb1 = *(const f32x4*)(bbase + 4);
  rb2 = *(const f32x4*)(bbase + ldb);
  rb3 = *(const f32x4*)(bbase + ldb + 4);
  {
    u32x4 w;
    w[0] = pk2(ra0.x * rsc, ra0.y * rsc);
    w[1] = pk2(ra0.z * rsc, ra0.w * rsc);
    w[2] = pk2(ra1.x * rsc, ra1.y * rsc);
    w[3] = pk2(ra1.z * rsc, ra1.w * rsc);
    *(u32x4*)&As[0][aRow][aCol] = w;
    #pragma unroll
    for (int j = 0; j < 8; ++j) {
      float lo = (j < 4) ? rb0[j] : rb1[j - 4];
      float hi = (j < 4) ? rb2[j] : rb3[j - 4];
      *(unsigned*)&Bs[0][bN8 + j][2 * bKp] = pk2(lo, hi);
    }
  }
  __syncthreads();

  for (int s = 0; s < nk; ++s) {
    const int cur = s & 1;
    const bool more = (s + 1) < nk;
    if (more) {  // issue next tile's global loads before computing
      int k0 = (s + 1) * 32;
      ra0 = *(const f32x4*)(ap + k0);
      ra1 = *(const f32x4*)(ap + k0 + 4);
      const float* bp = bbase + (size_t)k0 * ldb;
      rb0 = *(const f32x4*)(bp + 0);
      rb1 = *(const f32x4*)(bp + 4);
      rb2 = *(const f32x4*)(bp + ldb);
      rb3 = *(const f32x4*)(bp + ldb + 4);
    }

    FragAB af[2], bf[2];
    #pragma unroll
    for (int i = 0; i < 2; ++i) {
      int row = wm * 32 + i * 16 + l16;
      af[i].q[0] = *(const u32x4*)&As[cur][row][hs * 8];
      af[i].q[1] = *(const u32x4*)&As[cur][row][16 + hs * 8];
    }
    #pragma unroll
    for (int j = 0; j < 2; ++j) {
      int col = wn * 32 + j * 16 + l16;
      bf[j].q[0] = *(const u32x4*)&Bs[cur][col][hs * 8];
      bf[j].q[1] = *(const u32x4*)&Bs[cur][col][16 + hs * 8];
    }
    #pragma unroll
    for (int i = 0; i < 2; ++i)
      #pragma unroll
      for (int j = 0; j < 2; ++j)
        acc[i][j] = __builtin_amdgcn_wmma_f32_16x16x32_bf16(
            false, af[i].v, false, bf[j].v, (short)0, acc[i][j], false, false);

    if (more) {  // stage next tile into the other buffer
      const int nxt = cur ^ 1;
      u32x4 w;
      w[0] = pk2(ra0.x * rsc, ra0.y * rsc);
      w[1] = pk2(ra0.z * rsc, ra0.w * rsc);
      w[2] = pk2(ra1.x * rsc, ra1.y * rsc);
      w[3] = pk2(ra1.z * rsc, ra1.w * rsc);
      *(u32x4*)&As[nxt][aRow][aCol] = w;
      #pragma unroll
      for (int j = 0; j < 8; ++j) {
        float lo = (j < 4) ? rb0[j] : rb1[j - 4];
        float hi = (j < 4) ? rb2[j] : rb3[j - 4];
        *(unsigned*)&Bs[nxt][bN8 + j][2 * bKp] = pk2(lo, hi);
      }
    }
    __syncthreads();
  }

  #pragma unroll
  for (int i = 0; i < 2; ++i)
    #pragma unroll
    for (int j = 0; j < 2; ++j) {
      int col = n0 + wn * 32 + j * 16 + l16;
      if (col >= N) continue;
      float bv = (bias != nullptr) ? bias[col] : 0.0f;
      #pragma unroll
      for (int v = 0; v < 8; ++v) {
        int row = m0 + wm * 32 + i * 16 + hs * 8 + v;
        if (row >= M) continue;
        float val = acc[i][j][v] + bv;
        if (flags & 2) val = fmaxf(val, 0.0f);
        size_t idx = (size_t)row * ldc + col;
        if (flags & 1) C[idx] += val;
        else           C[idx] = val;
      }
    }
}

// ---------------------------------------------------------------------------
// Flash attention with distance / log-delta bias, causal mask, and per-head
// struct accumulation (attn @ delta). 128 threads: wave w owns q-tile w.
// ---------------------------------------------------------------------------
__global__ __launch_bounds__(128) void attn_kernel(
    const float* __restrict__ Q, const float* __restrict__ Kmat,
    const float* __restrict__ Vmat, const float* __restrict__ delta,
    float* __restrict__ ctxOut, float* __restrict__ structOut)
{
  __shared__ __attribute__((aligned(16))) unsigned short Aq[64 * 64];
  __shared__ __attribute__((aligned(16))) unsigned short Kt[64 * 64];
  __shared__ __attribute__((aligned(16))) unsigned short Vt[64 * 64];
  __shared__ __attribute__((aligned(16))) unsigned short Pl[4][16 * 64];
  __shared__ float deltaS[64];
  __shared__ float logdS[64];

  const int qb   = blockIdx.x;
  const int head = blockIdx.y;
  const int b    = blockIdx.z;
  const int tid  = threadIdx.x;
  const int lane = tid & 31;
  const int qt   = tid >> 5;
  const int hs   = lane >> 4;
  const int l16  = lane & 15;
  const float scale = 0.125f;  // 1/sqrt(64)

  const int srow = tid >> 1;            // staging: row, 32-elem half
  const int sc0  = (tid & 1) * 32;
  const int vkey = (tid & 31) * 2;      // V staging: key pair
  const int vc0  = (tid >> 5) * 16;     // V staging: 16-d chunk

  // stage Q block (vectorized, packed)
  {
    const float* qp = Q + ((size_t)(b * LSEQ + qb * 64 + srow)) * D_MODEL + head * HDIM + sc0;
    #pragma unroll
    for (int j2 = 0; j2 < 4; ++j2) {
      f32x4 va = *(const f32x4*)(qp + j2 * 8);
      f32x4 vb = *(const f32x4*)(qp + j2 * 8 + 4);
      u32x4 w;
      w[0] = pk2(va.x, va.y); w[1] = pk2(va.z, va.w);
      w[2] = pk2(vb.x, vb.y); w[3] = pk2(vb.z, vb.w);
      *(u32x4*)&Aq[srow * 64 + sc0 + j2 * 8] = w;
    }
  }
  __syncthreads();

  FragAB aq[2];
  {
    int row = qt * 16 + l16;
    #pragma unroll
    for (int kc = 0; kc < 2; ++kc) {
      aq[kc].q[0] = *(const u32x4*)&Aq[row * 64 + kc * 32 + hs * 8];
      aq[kc].q[1] = *(const u32x4*)&Aq[row * 64 + kc * 32 + 16 + hs * 8];
    }
  }

  float mrun[8], den[8], snum[8];
  v8f cacc[4];
  #pragma unroll
  for (int v = 0; v < 8; ++v) { mrun[v] = -3.0e38f; den[v] = 0.0f; snum[v] = 0.0f; }
  #pragma unroll
  for (int d = 0; d < 4; ++d) cacc[d] = vzero8();

  for (int kb = 0; kb <= qb; ++kb) {
    __syncthreads();
    {
      // K block: natural layout, vectorized packed stores
      const float* kp = Kmat + ((size_t)(b * LSEQ + kb * 64 + srow)) * D_MODEL
                             + head * HDIM + sc0;
      #pragma unroll
      for (int j2 = 0; j2 < 4; ++j2) {
        f32x4 va = *(const f32x4*)(kp + j2 * 8);
        f32x4 vb = *(const f32x4*)(kp + j2 * 8 + 4);
        u32x4 w;
        w[0] = pk2(va.x, va.y); w[1] = pk2(va.z, va.w);
        w[2] = pk2(vb.x, vb.y); w[3] = pk2(vb.z, vb.w);
        *(u32x4*)&Kt[srow * 64 + sc0 + j2 * 8] = w;
      }
      // V block: transposed; pack key pairs into dwords
      const float* vp0 = Vmat + ((size_t)(b * LSEQ + kb * 64 + vkey)) * D_MODEL
                              + head * HDIM + vc0;
      const float* vp1 = vp0 + D_MODEL;
      #pragma unroll
      for (int j4 = 0; j4 < 4; ++j4) {
        f32x4 v0 = *(const f32x4*)(vp0 + j4 * 4);
        f32x4 v1 = *(const f32x4*)(vp1 + j4 * 4);
        #pragma unroll
        for (int j = 0; j < 4; ++j)
          *(unsigned*)&Vt[(vc0 + j4 * 4 + j) * 64 + vkey] = pk2(v0[j], v1[j]);
      }
      if (tid < 64) {
        float dv = delta[(size_t)b * LSEQ + kb * 64 + tid];
        deltaS[tid] = dv;
        logdS[tid] = __logf(dv + 1e-6f);
      }
    }
    __syncthreads();

    v8f sf[4];
    #pragma unroll
    for (int nt = 0; nt < 4; ++nt) {
      sf[nt] = vzero8();
      #pragma unroll
      for (int kc = 0; kc < 2; ++kc) {
        FragAB bk;
        int col = nt * 16 + l16;
        bk.q[0] = *(const u32x4*)&Kt[col * 64 + kc * 32 + hs * 8];
        bk.q[1] = *(const u32x4*)&Kt[col * 64 + kc * 32 + 16 + hs * 8];
        sf[nt] = __builtin_amdgcn_wmma_f32_16x16x32_bf16(
            false, aq[kc].v, false, bk.v, (short)0, sf[nt], false, false);
      }
    }

    float mx[8];
    #pragma unroll
    for (int v = 0; v < 8; ++v) mx[v] = -3.0e38f;
    const int qbase = qb * 64 + qt * 16 + hs * 8;
    #pragma unroll
    for (int nt = 0; nt < 4; ++nt) {
      int kg = kb * 64 + nt * 16 + l16;
      float ld = logdS[nt * 16 + l16];
      #pragma unroll
      for (int v = 0; v < 8; ++v) {
        int qg = qbase + v;
        float s = (kg <= qg) ? sf[nt][v] * scale - 0.1f * (float)(qg - kg) + ld : -3.0e38f;
        sf[nt][v] = s;
        mx[v] = fmaxf(mx[v], s);
      }
    }
    #pragma unroll
    for (int off = 1; off < 16; off <<= 1)
      #pragma unroll
      for (int v = 0; v < 8; ++v)
        mx[v] = fmaxf(mx[v], __shfl_xor(mx[v], off, 32));

    float alpha[8], rs[8], ss[8];
    #pragma unroll
    for (int v = 0; v < 8; ++v) {
      float mnew = fmaxf(mrun[v], mx[v]);
      alpha[v] = __expf(mrun[v] - mnew);
      mrun[v] = mnew;
      rs[v] = 0.0f; ss[v] = 0.0f;
    }
    #pragma unroll
    for (int nt = 0; nt < 4; ++nt) {
      float dv = deltaS[nt * 16 + l16];
      #pragma unroll
      for (int v = 0; v < 8; ++v) {
        float e = __expf(sf[nt][v] - mrun[v]);
        sf[nt][v] = e;
        rs[v] += e;
        ss[v] += e * dv;
      }
    }
    #pragma unroll
    for (int off = 1; off < 16; off <<= 1)
      #pragma unroll
      for (int v = 0; v < 8; ++v) {
        rs[v] += __shfl_xor(rs[v], off, 32);
        ss[v] += __shfl_xor(ss[v], off, 32);
      }
    #pragma unroll
    for (int v = 0; v < 8; ++v) {
      den[v]  = den[v]  * alpha[v] + rs[v];
      snum[v] = snum[v] * alpha[v] + ss[v];
    }
    #pragma unroll
    for (int d = 0; d < 4; ++d)
      #pragma unroll
      for (int v = 0; v < 8; ++v) cacc[d][v] *= alpha[v];

    #pragma unroll
    for (int nt = 0; nt < 4; ++nt)
      #pragma unroll
      for (int v = 0; v < 8; ++v)
        Pl[qt][(hs * 8 + v) * 64 + nt * 16 + l16] = f2bf(sf[nt][v]);

    #pragma unroll
    for (int kc = 0; kc < 2; ++kc) {
      FragAB pa;
      pa.q[0] = *(const u32x4*)&Pl[qt][l16 * 64 + kc * 32 + hs * 8];
      pa.q[1] = *(const u32x4*)&Pl[qt][l16 * 64 + kc * 32 + 16 + hs * 8];
      #pragma unroll
      for (int dt = 0; dt < 4; ++dt) {
        FragAB bv;
        int dcol = dt * 16 + l16;
        bv.q[0] = *(const u32x4*)&Vt[dcol * 64 + kc * 32 + hs * 8];
        bv.q[1] = *(const u32x4*)&Vt[dcol * 64 + kc * 32 + 16 + hs * 8];
        cacc[dt] = __builtin_amdgcn_wmma_f32_16x16x32_bf16(
            false, pa.v, false, bv.v, (short)0, cacc[dt], false, false);
      }
    }
  }

  #pragma unroll
  for (int v = 0; v < 8; ++v) {
    int qg = qb * 64 + qt * 16 + hs * 8 + v;
    float inv = 1.0f / den[v];
    #pragma unroll
    for (int dt = 0; dt < 4; ++dt)
      ctxOut[((size_t)(b * LSEQ + qg)) * D_MODEL + head * HDIM + dt * 16 + l16] =
          cacc[dt][v] * inv;
    if (l16 == 0)
      structOut[((size_t)(b * NHEAD + head)) * LSEQ + qg] = snum[v] * inv;
  }
}

// ---------------------------------------------------------------------------
// Small fp32 kernels
// ---------------------------------------------------------------------------
__global__ __launch_bounds__(512) void ufeat1_kernel(const float* __restrict__ u_i,
    const float* __restrict__ w, const float* __restrict__ bias, float* __restrict__ out)
{
  int b = blockIdx.x, j = threadIdx.x;
  float s = bias[j];
  for (int i = 0; i < DUSER; ++i) s += u_i[b * DUSER + i] * w[i * D_MODEL + j];
  out[b * D_MODEL + j] = (s > 0.0f) ? s : 0.2f * s;  // leaky relu 0.2
}

__global__ __launch_bounds__(512) void ufeat2_kernel(const float* __restrict__ x,
    const float* __restrict__ w, const float* __restrict__ bias, float* __restrict__ out)
{
  int b = blockIdx.x, j = threadIdx.x;
  float s = bias[j];
  for (int i = 0; i < D_MODEL; ++i) s += x[b * D_MODEL + i] * w[i * D_MODEL + j];
  out[b * D_MODEL + j] = s;
}

__global__ __launch_bounds__(128) void embed_kernel(const int* __restrict__ c,
    const int* __restrict__ r, const float* __restrict__ t,
    const float* __restrict__ cemb, const float* __restrict__ remb,
    const float* __restrict__ pemb, const float* __restrict__ ufeat,
    float* __restrict__ h)
{
  int m = blockIdx.x;
  int b = m / LSEQ, l = m % LSEQ;
  int ci = c[m], ri = r[m];
  float tv = t[m];
  const float kdiv = -0.03597711149f;  // -ln(10000)/256
  #pragma unroll
  for (int i = 0; i < 4; ++i) {
    int d = threadIdx.x + i * 128;
    float te = (d < 256) ? __sinf(tv * __expf((float)d * kdiv))
                         : __cosf(tv * __expf((float)(d - 256) * kdiv));
    h[(size_t)m * D_MODEL + d] = cemb[(size_t)ci * D_MODEL + d]
                               + remb[(size_t)ri * D_MODEL + d]
                               + pemb[(size_t)l * D_MODEL + d]
                               + te + ufeat[b * D_MODEL + d];
  }
}

__global__ void pack_conv_kernel(const float* __restrict__ conv_w, float* __restrict__ packed)
{
  int idx = blockIdx.x * blockDim.x + threadIdx.x;
  if (idx >= D_MODEL * D_MODEL * KCONV) return;
  int o    = idx / (D_MODEL * KCONV);
  int rest = idx % (D_MODEL * KCONV);
  int in   = rest / KCONV;
  int k    = rest % KCONV;
  packed[(size_t)k * D_MODEL * D_MODEL + (size_t)in * D_MODEL + o] = conv_w[idx];
}

__device__ __forceinline__ float block_sum_128(float s, float* red)
{
  int lane = threadIdx.x & 31, wid = threadIdx.x >> 5;
  #pragma unroll
  for (int off = 16; off; off >>= 1) s += __shfl_xor(s, off, 32);
  __syncthreads();
  if (lane == 0) red[wid] = s;
  __syncthreads();
  return red[0] + red[1] + red[2] + red[3];
}

__global__ __launch_bounds__(128) void smooth_ln_kernel(float* __restrict__ h,
    const float* __restrict__ trend, const float* __restrict__ sqb,
    const float* __restrict__ g, const float* __restrict__ be)
{
  __shared__ float red[4];
  size_t base = (size_t)blockIdx.x * D_MODEL;
  float x[4];
  float s = 0.0f;
  #pragma unroll
  for (int i = 0; i < 4; ++i) {
    int d = threadIdx.x + i * 128;
    float hv = h[base + d], tv = trend[base + d], sq = sqb[d];
    float xv = tv + sq * sq * (hv - tv) + hv;
    x[i] = xv; s += xv;
  }
  float mean = block_sum_128(s, red) * (1.0f / D_MODEL);
  float vs = 0.0f;
  #pragma unroll
  for (int i = 0; i < 4; ++i) { x[i] -= mean; vs += x[i] * x[i]; }
  float var = block_sum_128(vs, red) * (1.0f / D_MODEL);
  float inv = rsqrtf(var + 1e-12f);
  #pragma unroll
  for (int i = 0; i < 4; ++i) {
    int d = threadIdx.x + i * 128;
    h[base + d] = g[d] * x[i] * inv + be[d];
  }
}

__global__ __launch_bounds__(128) void ln_add_kernel(float* __restrict__ h,
    const float* __restrict__ add, const float* __restrict__ g,
    const float* __restrict__ be, float eps)
{
  __shared__ float red[4];
  size_t base = (size_t)blockIdx.x * D_MODEL;
  float x[4];
  float s = 0.0f;
  #pragma unroll
  for (int i = 0; i < 4; ++i) {
    int d = threadIdx.x + i * 128;
    float xv = h[base + d] + add[base + d];
    x[i] = xv; s += xv;
  }
  float mean = block_sum_128(s, red) * (1.0f / D_MODEL);
  float vs = 0.0f;
  #pragma unroll
  for (int i = 0; i < 4; ++i) { x[i] -= mean; vs += x[i] * x[i]; }
  float var = block_sum_128(vs, red) * (1.0f / D_MODEL);
  float inv = rsqrtf(var + eps);
  #pragma unroll
  for (int i = 0; i < 4; ++i) {
    int d = threadIdx.x + i * 128;
    h[base + d] = g[d] * x[i] * inv + be[d];
  }
}

__global__ __launch_bounds__(256) void rp2_kernel(const float* __restrict__ tmp,
    const float* __restrict__ w2, const float* __restrict__ b2,
    float* __restrict__ delta0, float* __restrict__ deltaC)
{
  int lane = threadIdx.x & 31, wid = threadIdx.x >> 5;
  int row = blockIdx.x * 8 + wid;
  float s = 0.0f;
  #pragma unroll
  for (int j = 0; j < 8; ++j) {
    int i = lane * 8 + j;
    s += tmp[(size_t)row * 256 + i] * w2[i];
  }
  #pragma unroll
  for (int off = 16; off; off >>= 1) s += __shfl_xor(s, off, 32);
  if (lane == 0) {
    float d0 = 1.0f / (1.0f + __expf(-(s + b2[0])));
    delta0[row] = d0;
    deltaC[row] = d0;
  }
}

__global__ __launch_bounds__(64) void gate_kernel(const float* __restrict__ u_i,
    const float* __restrict__ w, const float* __restrict__ bias, float* __restrict__ gw)
{
  __shared__ float lg[8][8];
  int b = threadIdx.x >> 3, k = threadIdx.x & 7;
  float s = bias[k];
  for (int i = 0; i < DUSER; ++i) s += u_i[b * DUSER + i] * w[i * KEXP + k];
  lg[b][k] = s;
  __syncthreads();
  float mx = -3.0e38f;
  for (int j = 0; j < 8; ++j) mx = fmaxf(mx, lg[b][j]);
  float sum = 0.0f;
  for (int j = 0; j < 8; ++j) sum += __expf(lg[b][j] - mx);
  gw[b * KEXP + k] = __expf(lg[b][k] - mx) / sum;
}

__global__ void delta_update_kernel(const float* __restrict__ structW,
    const float* __restrict__ delta0, float* __restrict__ deltaC)
{
  int m = blockIdx.x * blockDim.x + threadIdx.x;
  if (m >= MROWS) return;
  int b = m / LSEQ, l = m % LSEQ;
  float s = 0.0f;
  #pragma unroll
  for (int hh = 0; hh < NHEAD; ++hh)
    s += structW[((size_t)(b * NHEAD + hh)) * LSEQ + l];
  deltaC[m] = 0.5f * (s * (1.0f / NHEAD)) + 0.5f * delta0[m];
}

__global__ __launch_bounds__(256) void ypred_kernel(const float* __restrict__ h,
    const float* __restrict__ exp_w, const float* __restrict__ exp_b,
    const float* __restrict__ gw, float* __restrict__ y)
{
  int lane = threadIdx.x & 31, wid = threadIdx.x >> 5;
  int row = blockIdx.x * 8 + wid;
  int b = row / LSEQ;
  float acc[KEXP];
  #pragma unroll
  for (int e = 0; e < KEXP; ++e) acc[e] = 0.0f;
  #pragma unroll
  for (int j = 0; j < 16; ++j) {
    int d = lane * 16 + j;
    float hv = h[(size_t)row * D_MODEL + d];
    #pragma unroll
    for (int e = 0; e < KEXP; ++e) acc[e] += hv * exp_w[e * D_MODEL + d];
  }
  #pragma unroll
  for (int off = 16; off; off >>= 1)
    #pragma unroll
    for (int e = 0; e < KEXP; ++e) acc[e] += __shfl_xor(acc[e], off, 32);
  if (lane == 0) {
    float s = 0.0f;
    #pragma unroll
    for (int e = 0; e < KEXP; ++e)
      s += gw[b * KEXP + e] / (1.0f + __expf(-(acc[e] + exp_b[e])));
    y[row] = s;
  }
}

// ---------------------------------------------------------------------------
extern "C" void kernel_launch(void* const* d_in, const int* in_sizes, int n_in,
                              void* d_out, int out_size, void* d_ws, size_t ws_size,
                              hipStream_t stream)
{
  (void)in_sizes; (void)n_in; (void)out_size; (void)ws_size;

  const int*   c      = (const int*)d_in[0];
  const int*   r      = (const int*)d_in[1];
  const float* t      = (const float*)d_in[2];
  const float* u_i    = (const float*)d_in[3];
  const float* cemb   = (const float*)d_in[4];
  const float* remb   = (const float*)d_in[5];
  const float* pemb   = (const float*)d_in[6];
  const float* ua_w1  = (const float*)d_in[7];
  const float* ua_b1  = (const float*)d_in[8];
  const float* ua_w2  = (const float*)d_in[9];
  const float* ua_b2  = (const float*)d_in[10];
  const float* conv_w = (const float*)d_in[11];
  const float* conv_b = (const float*)d_in[12];
  const float* sm_g   = (const float*)d_in[13];
  const float* sm_b   = (const float*)d_in[14];
  const float* sqb    = (const float*)d_in[15];
  const float* rp_w1  = (const float*)d_in[16];
  const float* rp_b1  = (const float*)d_in[17];
  const float* rp_w2  = (const float*)d_in[18];
  const float* rp_b2  = (const float*)d_in[19];
  const float* gate_w = (const float*)d_in[20];
  const float* gate_b = (const float*)d_in[21];
  const float* exp_w  = (const float*)d_in[22];
  const float* exp_b  = (const float*)d_in[23];

  float* out      = (float*)d_out;
  float* ypred    = out;
  float* deltaOut = out + MROWS;
  float* h        = out + 2 * MROWS;   // h lives in the output h-region

  float* ws      = (float*)d_ws;
  float* regA    = ws;                 // 16,777,216 floats (64 MB)
  float* regB    = ws + 16777216;      //  4,194,304 floats (16 MB)
  float* structW = ws + 20971520;      // 65,536
  float* delta0  = ws + 21037056;      // 8,192
  float* deltaC  = ws + 21045248;      // 8,192
  float* tmpU    = ws + 21053440;      // 4,096
  float* ufeat   = ws + 21057536;      // 4,096
  float* gw      = ws + 21061632;      // 64
  float* packW   = regA;               // 5*512*512 (early phase only)
  float* tmp256  = regA + 2097152;     // 8192*256  (early phase only)
  float* trend   = regB;

  // --- user feature MLP + embeddings -------------------------------------
  ufeat1_kernel<<<dim3(BATCH), dim3(512), 0, stream>>>(u_i, ua_w1, ua_b1, tmpU);
  ufeat2_kernel<<<dim3(BATCH), dim3(512), 0, stream>>>(tmpU, ua_w2, ua_b2, ufeat);
  embed_kernel<<<dim3(MROWS), dim3(128), 0, stream>>>(c, r, t, cemb, remb, pemb, ufeat, h);

  // --- causal channel-mixing conv as 5 shifted accumulating GEMMs --------
  pack_conv_kernel<<<dim3((D_MODEL * D_MODEL * KCONV + 255) / 256), dim3(256), 0, stream>>>(
      conv_w, packW);
  for (int kc = 0; kc < KCONV; ++kc) {
    gemm_wmma<<<dim3(D_MODEL / 128, MROWS / 64), dim3(256), 0, stream>>>(
        h, D_MODEL, kc - (KCONV - 1), LSEQ, nullptr,
        packW + (size_t)kc * D_MODEL * D_MODEL, D_MODEL,
        trend, D_MODEL, (kc == 0) ? conv_b : nullptr,
        MROWS, D_MODEL, D_MODEL, (kc == 0) ? 0 : 1);
  }
  smooth_ln_kernel<<<dim3(MROWS), dim3(128), 0, stream>>>(h, trend, sqb, sm_g, sm_b);

  // --- reliability predictor + gate --------------------------------------
  gemm_wmma<<<dim3(2, MROWS / 64), dim3(256), 0, stream>>>(
      h, D_MODEL, 0, 0, nullptr, rp_w1, 256, tmp256, 256, rp_b1,
      MROWS, 256, D_MODEL, 2 /*relu*/);
  rp2_kernel<<<dim3(MROWS / 8), dim3(256), 0, stream>>>(tmp256, rp_w2, rp_b2, delta0, deltaC);
  gate_kernel<<<dim3(1), dim3(64), 0, stream>>>(u_i, gate_w, gate_b, gw);

  float* qbuf = regA;
  float* kbuf = regA + 4194304;
  float* vbuf = regA + 8388608;
  float* ctx  = regA + 12582912;
  float* ff   = regA;                  // reused after attention consumed

  for (int layer = 0; layer < 2; ++layer) {
    int base = 24 + layer * 16;
    const float* q_w  = (const float*)d_in[base + 0];
    const float* q_b  = (const float*)d_in[base + 1];
    const float* k_w  = (const float*)d_in[base + 2];
    const float* k_b  = (const float*)d_in[base + 3];
    const float* v_w  = (const float*)d_in[base + 4];
    const float* v_b  = (const float*)d_in[base + 5];
    const float* o_w  = (const float*)d_in[base + 6];
    const float* o_b  = (const float*)d_in[base + 7];
    const float* n1_g = (const float*)d_in[base + 8];
    const float* n1_b = (const float*)d_in[base + 9];
    const float* n2_g = (const float*)d_in[base + 10];
    const float* n2_b = (const float*)d_in[base + 11];
    const float* f_w1 = (const float*)d_in[base + 12];
    const float* f_b1 = (const float*)d_in[base + 13];
    const float* f_w2 = (const float*)d_in[base + 14];
    const float* f_b2 = (const float*)d_in[base + 15];

    gemm_wmma<<<dim3(4, MROWS / 64), dim3(256), 0, stream>>>(
        h, D_MODEL, 0, 0, nullptr, q_w, D_MODEL, qbuf, D_MODEL, q_b,
        MROWS, D_MODEL, D_MODEL, 0);
    gemm_wmma<<<dim3(4, MROWS / 64), dim3(256), 0, stream>>>(
        h, D_MODEL, 0, 0, nullptr, k_w, D_MODEL, kbuf, D_MODEL, k_b,
        MROWS, D_MODEL, D_MODEL, 0);
    gemm_wmma<<<dim3(4, MROWS / 64), dim3(256), 0, stream>>>(
        h, D_MODEL, 0, 0, deltaC, v_w, D_MODEL, vbuf, D_MODEL, v_b,
        MROWS, D_MODEL, D_MODEL, 0);

    attn_kernel<<<dim3(LSEQ / 64, NHEAD, BATCH), dim3(128), 0, stream>>>(
        qbuf, kbuf, vbuf, deltaC, ctx, structW);

    gemm_wmma<<<dim3(4, MROWS / 64), dim3(256), 0, stream>>>(
        ctx, D_MODEL, 0, 0, nullptr, o_w, D_MODEL, regB, D_MODEL, o_b,
        MROWS, D_MODEL, D_MODEL, 0);
    ln_add_kernel<<<dim3(MROWS), dim3(128), 0, stream>>>(h, regB, n1_g, n1_b, 1e-5f);

    gemm_wmma<<<dim3(DFF / 128, MROWS / 64), dim3(256), 0, stream>>>(
        h, D_MODEL, 0, 0, nullptr, f_w1, DFF, ff, DFF, f_b1,
        MROWS, DFF, D_MODEL, 2 /*relu*/);
    gemm_wmma<<<dim3(4, MROWS / 64), dim3(256), 0, stream>>>(
        ff, DFF, 0, 0, nullptr, f_w2, D_MODEL, regB, D_MODEL, f_b2,
        MROWS, D_MODEL, DFF, 0);
    ln_add_kernel<<<dim3(MROWS), dim3(128), 0, stream>>>(h, regB, n2_g, n2_b, 1e-5f);

    delta_update_kernel<<<dim3((MROWS + 255) / 256), dim3(256), 0, stream>>>(
        structW, delta0, deltaC);
  }

  ypred_kernel<<<dim3(MROWS / 8), dim3(256), 0, stream>>>(h, exp_w, exp_b, gw, ypred);
  hipMemcpyAsync(deltaOut, deltaC, MROWS * sizeof(float),
                 hipMemcpyDeviceToDevice, stream);
}
